// MultiHeadAttention_7825430413525
// MI455X (gfx1250) — matrix operand order repaired and checked by reading, not compiled, not run on hardware
//
#include <hip/hip_runtime.h>
#include <hip/hip_bf16.h>

// MHA forward for MI455X (gfx1250, wave32, WMMA).
// B=4, S=2048, D=1024, H=16, dk=64.  ~137 GFLOP; mandatory HBM traffic is
// dominated by the 1.07 GB attn output. All matmuls run on the bf16 WMMA pipe
// (f32 accumulate). Projections use an LDS-tiled 128x128-per-block GEMM
// (double-buffered bf16 panels) for 8x/4x register-level reuse; attention
// keeps the SxS scores register-resident per q-tile and writes attn once.

#define DMODEL 1024
#define NHEADS 16
#define DK     64
#define BATCH  4
#define SEQ    2048

#define LDP 40  // LDS panel row pitch in halfwords (80B: 16B-aligned, bank-spread)

typedef __bf16 bf16_t;
typedef __attribute__((ext_vector_type(16))) __bf16 v16bf;
typedef __attribute__((ext_vector_type(8)))  float  v8f;

__device__ __forceinline__ bf16_t f2bf(float x) { return (bf16_t)x; }

__device__ __forceinline__ v8f wmma_bf16(v16bf a, v16bf b, v8f c) {
  // D = A(16x32 bf16) * B(32x16 bf16) + C(16x16 f32)
  return __builtin_amdgcn_wmma_f32_16x16x32_bf16(false, a, false, b, (short)0, c,
                                                 false, false);
}

// Stage a 128-row x 32-k panel (f32 or bf16 source) into LDS as bf16.
template <typename T>
__device__ __forceinline__ void stage_panel(const T* __restrict__ src, size_t rowBase,
                                            int k0, int Kstride,
                                            bf16_t* __restrict__ dst, int tid) {
#pragma unroll
  for (int it = 0; it < 4; ++it) {
    const int idx = tid + it * 256;   // 0..1023 -> 128 rows x 8 groups of 4
    const int r   = idx >> 3;
    const int kg  = (idx & 7) << 2;
    const T* s = src + (rowBase + r) * (size_t)Kstride + k0 + kg;
    bf16_t* d  = dst + r * LDP + kg;
#pragma unroll
    for (int j = 0; j < 4; ++j) d[j] = f2bf((float)s[j]);
  }
}

// ---------------------------------------------------------------------------
// Tiled GEMM with bias: out[M,N] = A[M,K] @ W[N,K]^T + bias[N]
// Block = 256 threads (8 waves) -> 128x128 output tile.
// Wave (wm,wn) = (wave>>1, wave&1) computes a 32x64 region = 2x4 WMMA tiles.
// K-loop in steps of 32 with double-buffered bf16 LDS panels.
// ---------------------------------------------------------------------------
template <typename AT, typename OT>
__global__ __launch_bounds__(256)
void gemm_bias_wmma(const AT* __restrict__ A, const float* __restrict__ W,
                    const float* __restrict__ bias, OT* __restrict__ out,
                    int M, int N, int K) {
  __shared__ bf16_t ldsA[2][128 * LDP];
  __shared__ bf16_t ldsB[2][128 * LDP];

  const int tid    = threadIdx.x;
  const int lane   = tid & 31;
  const int wave   = tid >> 5;
  const int lane16 = lane & 15;
  const int hi     = lane >> 4;
  const int wm     = wave >> 1;  // 0..3 -> 32-row slice
  const int wn     = wave & 1;   // 0..1 -> 64-col slice

  const size_t rowM = (size_t)blockIdx.y * 128;  // A rows
  const size_t rowN = (size_t)blockIdx.x * 128;  // W rows (output cols)

  v8f acc[2][4];
#pragma unroll
  for (int mi = 0; mi < 2; ++mi)
#pragma unroll
    for (int ni = 0; ni < 4; ++ni) acc[mi][ni] = (v8f){};

  const int ksteps = K >> 5;
  stage_panel(A, rowM, 0, K, ldsA[0], tid);
  stage_panel(W, rowN, 0, K, ldsB[0], tid);

  for (int ks = 0; ks < ksteps; ++ks) {
    __syncthreads();
    const int cur = ks & 1;

    if (ks + 2 < ksteps) {  // speculative prefetch two panels ahead (L2)
      const int idx = tid;
      const int r = idx >> 3, kg = (idx & 7) << 2;
      __builtin_prefetch(A + (rowM + r) * (size_t)K + (ks + 2) * 32 + kg, 0, 1);
      __builtin_prefetch(W + (rowN + r) * (size_t)K + (ks + 2) * 32 + kg, 0, 1);
    }
    if (ks + 1 < ksteps) {  // stage next panels into the other buffer
      stage_panel(A, rowM, (ks + 1) * 32, K, ldsA[cur ^ 1], tid);
      stage_panel(W, rowN, (ks + 1) * 32, K, ldsB[cur ^ 1], tid);
    }

    // A fragments: 2 M-subtiles of this wave
    v16bf af[2];
#pragma unroll
    for (int mi = 0; mi < 2; ++mi) {
      const bf16_t* ar = &ldsA[cur][(wm * 32 + mi * 16 + lane16) * LDP];
#pragma unroll
      for (int i = 0; i < 8; ++i) {
        af[mi][i]     = ar[hi * 8 + i];
        af[mi][i + 8] = ar[hi * 8 + 16 + i];
      }
    }
    // B fragments: 4 N-subtiles of this wave
    v16bf bf[4];
#pragma unroll
    for (int ni = 0; ni < 4; ++ni) {
      const bf16_t* br = &ldsB[cur][(wn * 64 + ni * 16 + lane16) * LDP];
#pragma unroll
      for (int i = 0; i < 8; ++i) {
        bf[ni][i]     = br[hi * 8 + i];
        bf[ni][i + 8] = br[hi * 8 + 16 + i];
      }
    }
#pragma unroll
    for (int mi = 0; mi < 2; ++mi)
#pragma unroll
      for (int ni = 0; ni < 4; ++ni)
        acc[mi][ni] = wmma_bf16(af[mi], bf[ni], acc[mi][ni]);
  }

  // Epilogue: bias + store
#pragma unroll
  for (int ni = 0; ni < 4; ++ni) {
    const int n  = (int)rowN + wn * 64 + ni * 16 + lane16;
    const float bn = bias[n];
#pragma unroll
    for (int mi = 0; mi < 2; ++mi)
#pragma unroll
      for (int vi = 0; vi < 8; ++vi) {
        const size_t m = rowM + wm * 32 + mi * 16 + vi + hi * 8;
        out[m * N + n] = (OT)(acc[mi][ni][vi] + bn);
      }
  }
}

// ---------------------------------------------------------------------------
// Fused attention for one (batch, head, 16-query tile) per 256-thread block.
// Wave w owns the 256-key column chunk [w*256, w*256+256).
// Scores stay in VGPRs (16 x v8f per lane); softmax via lane shuffles + tiny
// LDS; P is written once to global attn and bounced through 2KB of wave-
// private LDS to feed the P@V WMMA A-fragments.
// ---------------------------------------------------------------------------
__global__ __launch_bounds__(256)
void attn_fused(const bf16_t* __restrict__ qp, const bf16_t* __restrict__ kp,
                const bf16_t* __restrict__ vp, bf16_t* __restrict__ ctx,
                float* __restrict__ attn) {
  __shared__ float red[8][16];      // cross-wave row max / row sum
  __shared__ float smem[8 * 1024];  // per-wave: 2KB P-transpose, then 16x64 partial ctx

  const int tid    = threadIdx.x;
  const int lane   = tid & 31;
  const int wave   = tid >> 5;
  const int lane16 = lane & 15;
  const int hi     = lane >> 4;

  const int qt = blockIdx.x;  // 0..S/16-1
  const int h  = blockIdx.y;  // 0..H-1
  const int b  = blockIdx.z;  // 0..B-1

  const size_t rowbase = (size_t)b * SEQ;

  // ---- A fragments of Q (dk=64 -> two 16x32 k-steps), loaded once ----
  const bf16_t* qrow = qp + (rowbase + qt * 16 + lane16) * DMODEL + h * DK;
  v16bf aq0, aq1;
#pragma unroll
  for (int i = 0; i < 8; ++i) {
    aq0[i]     = qrow[hi * 8 + i];
    aq0[i + 8] = qrow[hi * 8 + 16 + i];
    aq1[i]     = qrow[32 + hi * 8 + i];
    aq1[i + 8] = qrow[32 + hi * 8 + 16 + i];
  }

  // ---- Phase A: scores = (Q K^T) * 1/sqrt(dk), register-resident ----
  v8f sc[16];
#pragma unroll
  for (int t = 0; t < 16; ++t) {
    const int jt = wave * 16 + t;
    const bf16_t* krow = kp + (rowbase + jt * 16 + lane16) * DMODEL + h * DK;
    v16bf bk0, bk1;
    const int kb = hi * 8;
#pragma unroll
    for (int i = 0; i < 8; ++i) {
      bk0[i]     = krow[kb + i];        // K columns 0..31 of this k-step
      bk0[i + 8] = krow[kb + 16 + i];
      bk1[i]     = krow[32 + kb + i];   // K columns 32..63
      bk1[i + 8] = krow[32 + kb + 16 + i];
    }
    v8f c = {};
    c = wmma_bf16(aq0, bk0, c);
    c = wmma_bf16(aq1, bk1, c);
#pragma unroll
    for (int vi = 0; vi < 8; ++vi) sc[t][vi] = c[vi] * 0.125f;  // 1/sqrt(64)
  }

  // ---- Phase B: softmax over the full 2048-wide rows ----
  float rmax[8];
#pragma unroll
  for (int vi = 0; vi < 8; ++vi) {
    float m = sc[0][vi];
#pragma unroll
    for (int t = 1; t < 16; ++t) m = fmaxf(m, sc[t][vi]);
#pragma unroll
    for (int o = 1; o < 16; o <<= 1) m = fmaxf(m, __shfl_xor(m, o, 32));
    rmax[vi] = m;
  }
  if (lane16 == 0) {
#pragma unroll
    for (int vi = 0; vi < 8; ++vi) red[wave][vi + hi * 8] = rmax[vi];
  }
  __syncthreads();
#pragma unroll
  for (int vi = 0; vi < 8; ++vi) {
    float m = red[0][vi + hi * 8];
#pragma unroll
    for (int w = 1; w < 8; ++w) m = fmaxf(m, red[w][vi + hi * 8]);
    rmax[vi] = m;
  }
  __syncthreads();  // red will be reused for sums

  float srow[8];
#pragma unroll
  for (int vi = 0; vi < 8; ++vi) srow[vi] = 0.f;
#pragma unroll
  for (int t = 0; t < 16; ++t)
#pragma unroll
    for (int vi = 0; vi < 8; ++vi) {
      float e = __expf(sc[t][vi] - rmax[vi]);
      sc[t][vi] = e;
      srow[vi] += e;
    }
#pragma unroll
  for (int vi = 0; vi < 8; ++vi)
#pragma unroll
    for (int o = 1; o < 16; o <<= 1) srow[vi] += __shfl_xor(srow[vi], o, 32);
  if (lane16 == 0) {
#pragma unroll
    for (int vi = 0; vi < 8; ++vi) red[wave][vi + hi * 8] = srow[vi];
  }
  __syncthreads();
  float rinv[8];
#pragma unroll
  for (int vi = 0; vi < 8; ++vi) {
    float s = 0.f;
#pragma unroll
    for (int w = 0; w < 8; ++w) s += red[w][vi + hi * 8];
    rinv[vi] = 1.f / s;
  }

  // normalize in registers + single write of the attn matrix
  float* attnbase = attn + ((size_t)(b * NHEADS + h) * SEQ + qt * 16) * SEQ;
#pragma unroll
  for (int t = 0; t < 16; ++t) {
    const int col = wave * 256 + t * 16 + lane16;
#pragma unroll
    for (int vi = 0; vi < 8; ++vi) {
      const float p = sc[t][vi] * rinv[vi];
      sc[t][vi] = p;
      attnbase[(size_t)(vi + hi * 8) * SEQ + col] = p;
    }
  }

  // ---- Phase C: ctx_partial = P_chunk @ V_chunk (16 x 64 per wave) ----
  float* pbuf = smem + wave * 1024;  // wave-private 16x32 staging (stride 32)
  v8f acc0 = {}, acc1 = {}, acc2 = {}, acc3 = {};
#pragma unroll
  for (int kk = 0; kk < 8; ++kk) {
    // stage 32 P-columns through LDS (C-layout -> A-fragment transpose)
#pragma unroll
    for (int t2 = 0; t2 < 2; ++t2) {
      const int t = kk * 2 + t2;
#pragma unroll
      for (int vi = 0; vi < 8; ++vi)
        pbuf[(vi + hi * 8) * 32 + t2 * 16 + lane16] = sc[t][vi];
    }
    v16bf ap;  // wave-private LDS: DS ops are in-order within a wave
#pragma unroll
    for (int i = 0; i < 8; ++i) {
      ap[i]     = f2bf(pbuf[lane16 * 32 + hi * 8 + i]);
      ap[i + 8] = f2bf(pbuf[lane16 * 32 + 16 + hi * 8 + i]);
    }
    const int krow0 = wave * 256 + kk * 32;
    const bf16_t* vbase = vp + (rowbase + krow0) * DMODEL + h * DK + lane16;
    v16bf bv0, bv1, bv2, bv3;
#pragma unroll
    for (int i = 0; i < 16; ++i) {
      const int kr = hi * 8 + (i & 7) + ((i >> 3) << 4);
      const bf16_t* vrow = vbase + (size_t)kr * DMODEL;
      bv0[i] = vrow[0];
      bv1[i] = vrow[16];
      bv2[i] = vrow[32];
      bv3[i] = vrow[48];
    }
    acc0 = wmma_bf16(ap, bv0, acc0);
    acc1 = wmma_bf16(ap, bv1, acc1);
    acc2 = wmma_bf16(ap, bv2, acc2);
    acc3 = wmma_bf16(ap, bv3, acc3);
  }

  // ---- reduce the 8 per-wave partials and emit bf16 ctx (B,S,D layout) ----
  __syncthreads();
  float* part = smem + wave * 1024;  // 16 x 64 f32
#pragma unroll
  for (int vi = 0; vi < 8; ++vi) {
    const int m = vi + hi * 8;
    part[m * 64 + 0  + lane16] = acc0[vi];
    part[m * 64 + 16 + lane16] = acc1[vi];
    part[m * 64 + 32 + lane16] = acc2[vi];
    part[m * 64 + 48 + lane16] = acc3[vi];
  }
  __syncthreads();
  for (int e = tid; e < 1024; e += 256) {
    const int m = e >> 6, d = e & 63;
    float s = 0.f;
#pragma unroll
    for (int w = 0; w < 8; ++w) s += smem[w * 1024 + m * 64 + d];
    ctx[(rowbase + qt * 16 + m) * DMODEL + h * DK + d] = f2bf(s);
  }
}

// ---------------------------------------------------------------------------
extern "C" void kernel_launch(void* const* d_in, const int* in_sizes, int n_in,
                              void* d_out, int out_size, void* d_ws, size_t ws_size,
                              hipStream_t stream) {
  const float* Q  = (const float*)d_in[0];
  const float* K  = (const float*)d_in[1];
  const float* V  = (const float*)d_in[2];
  const float* Wq = (const float*)d_in[3];
  const float* bq = (const float*)d_in[4];
  const float* Wk = (const float*)d_in[5];
  const float* bk = (const float*)d_in[6];
  const float* Wv = (const float*)d_in[7];
  const float* bv = (const float*)d_in[8];
  const float* Wo = (const float*)d_in[9];
  const float* bo = (const float*)d_in[10];

  float* out  = (float*)d_out;                          // (B,S,D)
  float* attn = out + (size_t)BATCH * SEQ * DMODEL;     // (B,H,S,S)

  // Workspace: q,k,v,ctx as bf16, 16 MB each = 64 MB total.
  char* ws = (char*)d_ws;
  const size_t projBytes = (size_t)BATCH * SEQ * DMODEL * sizeof(bf16_t);
  bf16_t* qws = (bf16_t*)(ws);
  bf16_t* kws = (bf16_t*)(ws + projBytes);
  bf16_t* vws = (bf16_t*)(ws + 2 * projBytes);
  bf16_t* cws = (bf16_t*)(ws + 3 * projBytes);

  const int M = BATCH * SEQ, N = DMODEL, Kd = DMODEL;
  const dim3 blk(256);
  const dim3 gproj(N / 128, M / 128);  // 128x128 tile per block

  gemm_bias_wmma<float, bf16_t><<<gproj, blk, 0, stream>>>(Q, Wq, bq, qws, M, N, Kd);
  gemm_bias_wmma<float, bf16_t><<<gproj, blk, 0, stream>>>(K, Wk, bk, kws, M, N, Kd);
  gemm_bias_wmma<float, bf16_t><<<gproj, blk, 0, stream>>>(V, Wv, bv, vws, M, N, Kd);

  const dim3 agrid(SEQ / 16, NHEADS, BATCH);
  attn_fused<<<agrid, blk, 0, stream>>>(qws, kws, vws, cws, attn);

  gemm_bias_wmma<bf16_t, float><<<gproj, blk, 0, stream>>>(cws, Wo, bo, out, M, N, Kd);
}
